// FRU_Adapter_8272107012296
// MI455X (gfx1250) — compile-verified, matrix operand order checked by reading, not run on hardware
//
#include <hip/hip_runtime.h>

#define BB 1576
#define TT 30
#define DD 1024
#define HH 128
#define BT (BB*TT)              // 47280
#define NBLK (BT/16)            // 2955
#define LN_EPS 1e-5f

typedef float  v2f   __attribute__((ext_vector_type(2)));
typedef float  v8f   __attribute__((ext_vector_type(8)));
typedef __bf16 bf16x4 __attribute__((ext_vector_type(4)));
typedef __bf16 bf16x8 __attribute__((ext_vector_type(8)));
typedef __bf16 v16bf  __attribute__((ext_vector_type(16)));

// ---------------------------------------------------------------------------
// bf16 WMMA 16x16 tile: C += A[16,K] * B[K,16], K multiple of 32.
// A: row-major bf16 with row stride lda (LDS). Per ISA 16-bit A layout:
//   lane m=lane&15, half=lane>>4; element e -> K = 32*kb + (e&7) + 16*(e>>3) + 8*half
// B: pre-swizzled so each lane reads 16 contiguous bf16 (32B) per k-block.
// ---------------------------------------------------------------------------
__device__ __forceinline__ v8f wmma_bf16_tile(const __bf16* A, int lda,
                                              const __bf16* Bsw, int bstride,
                                              int nkb, v8f acc) {
  const int lane = threadIdx.x & 31;
  const int hl   = lane >> 4;
  const int m    = lane & 15;
  for (int kb = 0; kb < nkb; ++kb) {
    const __bf16* ap = A + m * lda + kb * 32 + 8 * hl;
    bf16x8 alo = *(const bf16x8*)ap;
    bf16x8 ahi = *(const bf16x8*)(ap + 16);
    v16bf av = __builtin_shufflevector(alo, ahi, 0,1,2,3,4,5,6,7,8,9,10,11,12,13,14,15);
    const __bf16* bp = Bsw + kb * bstride + lane * 16;
    bf16x8 blo = *(const bf16x8*)bp;
    bf16x8 bhi = *(const bf16x8*)(bp + 8);
    v16bf bv = __builtin_shufflevector(blo, bhi, 0,1,2,3,4,5,6,7,8,9,10,11,12,13,14,15);
    acc = __builtin_amdgcn_wmma_f32_16x16x32_bf16(false, av, false, bv,
                                                  (short)0, acc, false, false);
  }
  return acc;
}

// ---------------------------------------------------------------------------
// f32 WMMA 16x16 tile (K multiple of 4) with generic strided B accessor.
// A layout (32-bit 16x4): lanes 0-15 K={k,k+1}, lanes 16-31 K={k+2,k+3}.
// B element (kk,n) at Bp[kk*ldbK + n*ldbN]  (supports transposed reads).
// ---------------------------------------------------------------------------
__device__ __forceinline__ v8f wmma_f32_tile(const float* A, int lda,
                                             const float* Bp, int ldbK, int ldbN,
                                             int K, v8f acc) {
  const int lane = threadIdx.x & 31;
  const int hl   = lane >> 4;
  const int lr   = lane & 15;
  for (int k0 = 0; k0 < K; k0 += 4) {
    int ka = k0 + 2 * hl;
    v2f a, b;
    a.x = A[lr * lda + ka];
    a.y = A[lr * lda + ka + 1];
    b.x = Bp[ka * ldbK + lr * ldbN];
    b.y = Bp[(ka + 1) * ldbK + lr * ldbN];
    acc = __builtin_amdgcn_wmma_f32_16x16x4_f32(false, a, false, b,
                                                (short)0, acc, false, false);
  }
  return acc;
}

// ---------------------------------------------------------------------------
// Pre-swizzle an f32 [K,N] weight into the per-lane bf16 WMMA-B layout:
// flat idx = ((kb*(N/16) + nt)*32 + lane)*16 + e
// ---------------------------------------------------------------------------
__global__ void swizzle_w(const float* __restrict__ W, __bf16* __restrict__ out,
                          int K, int N) {
  int idx = blockIdx.x * 256 + threadIdx.x;
  if (idx >= K * N) return;
  int e    = idx & 15;
  int lane = (idx >> 4) & 31;
  int rest = idx >> 9;
  int NT   = N >> 4;
  int nt   = rest % NT;
  int kb   = rest / NT;
  int hl   = lane >> 4;
  int n    = lane & 15;
  int k    = kb * 32 + (e & 7) + 16 * (e >> 3) + 8 * hl;
  out[idx] = (__bf16)W[(size_t)k * N + nt * 16 + n];
}

// ---------------------------------------------------------------------------
// Kernel 1: h = LN(x @ W1 + b1); gmean = mean_H(h)     [16 rows per block]
// ---------------------------------------------------------------------------
__global__ __launch_bounds__(256) void down_proj_ln(
    const float* __restrict__ x, const __bf16* __restrict__ W1b,
    const float* __restrict__ b1, const float* __restrict__ ln_g,
    const float* __restrict__ ln_b, float* __restrict__ h_ln,
    float* __restrict__ gmean) {
  __shared__ __align__(16) __bf16 xs[16 * 1032];   // padded stride: bank-safe
  __shared__ float hbuf[16 * 132];
  const int tid  = threadIdx.x;
  const int row0 = blockIdx.x * 16;

  // stage x slab [16,1024] as bf16
  for (int i = tid; i < (16 * 1024) / 4; i += 256) {
    int l = i * 4;
    int r = l >> 10, c = l & 1023;
    float4 v = *(const float4*)(x + (size_t)(row0 + r) * DD + c);
    bf16x4 bv;
    bv.x = (__bf16)v.x; bv.y = (__bf16)v.y;
    bv.z = (__bf16)v.z; bv.w = (__bf16)v.w;
    *(bf16x4*)(xs + r * 1032 + c) = bv;
  }
  __syncthreads();

  const int wave = tid >> 5, lane = tid & 31;
  const int hl = lane >> 4, lr = lane & 15;
  v8f acc = {};
  acc = wmma_bf16_tile(xs, 1032, W1b + wave * 512, 8 * 512, 32, acc);
  const float bias = b1[wave * 16 + lr];
#pragma unroll
  for (int i = 0; i < 8; ++i)
    hbuf[(i + 8 * hl) * 132 + wave * 16 + lr] = acc[i] + bias;
  __syncthreads();

  // fused LayerNorm + channel-mean (16 threads per row)
  const int r = tid >> 4, sub = tid & 15;
  float s = 0.f, ss = 0.f, vals[8];
#pragma unroll
  for (int j = 0; j < 8; ++j) {
    float v = hbuf[r * 132 + sub * 8 + j];
    vals[j] = v; s += v; ss += v * v;
  }
#pragma unroll
  for (int o = 8; o >= 1; o >>= 1) {
    s  += __shfl_xor(s, o, 16);
    ss += __shfl_xor(ss, o, 16);
  }
  float mean = s * (1.f / 128.f);
  float rstd = rsqrtf(ss * (1.f / 128.f) - mean * mean + LN_EPS);
  float gs = 0.f;
  float outv[8];
#pragma unroll
  for (int j = 0; j < 8; ++j) {
    int c = sub * 8 + j;
    float v = (vals[j] - mean) * rstd * ln_g[c] + ln_b[c];
    outv[j] = v;
    gs += v;
  }
  float* hp = h_ln + (size_t)(row0 + r) * HH + sub * 8;
  *(float4*)(hp)     = make_float4(outv[0], outv[1], outv[2], outv[3]);
  *(float4*)(hp + 4) = make_float4(outv[4], outv[5], outv[6], outv[7]);
#pragma unroll
  for (int o = 8; o >= 1; o >>= 1) gs += __shfl_xor(gs, o, 16);
  if (sub == 0) gmean[row0 + r] = gs * (1.f / 128.f);
}

// ---------------------------------------------------------------------------
// Kernel 2: per-batch FRU gate + single-head attention (T=30 padded to 32)
// Writes hcur as bf16 (it is only consumed by the bf16 up-projection GEMM).
// ---------------------------------------------------------------------------
__global__ __launch_bounds__(256) void batch_attn(
    const float* __restrict__ h_ln, const float* __restrict__ gmean,
    const float* __restrict__ T_W, const float* __restrict__ T_b,
    const __bf16* __restrict__ Wqb, const float* __restrict__ bq,
    const __bf16* __restrict__ Wkb, const float* __restrict__ bk,
    const __bf16* __restrict__ Wvb, const float* __restrict__ bv,
    const __bf16* __restrict__ Wob, const float* __restrict__ bo,
    const float* __restrict__ n_g, const float* __restrict__ n_b,
    __bf16* __restrict__ hcurb) {
  __shared__ float hg[32 * 136];
  __shared__ __align__(16) __bf16 hnb[32 * 136];
  __shared__ float qs[32 * 136], ks[32 * 136], vs[32 * 136];
  __shared__ float Ss[32 * 34];
  __shared__ __align__(16) __bf16 avb[32 * 136];
  __shared__ float gm[32], gate[32];

  const int b = blockIdx.x, tid = threadIdx.x;
  const int wave = tid >> 5, lane = tid & 31;
  const int hl = lane >> 4, lr = lane & 15;

  // temporal linear over mean-channel signal
  if (tid < TT) gm[tid] = gmean[b * TT + tid];
  __syncthreads();
  if (tid < TT) {
    float a = T_b[tid];
    for (int tp = 0; tp < TT; ++tp) a += gm[tp] * T_W[tp * TT + tid];
    gate[tid] = a;
  }
  if (tid >= TT && tid < 32) gate[tid] = -3.0e38f;
  __syncthreads();
  if (wave == 0) {  // softmax over T in one wave
    float v = gate[lane], m = v;
    for (int o = 16; o >= 1; o >>= 1) m = fmaxf(m, __shfl_xor(m, o, 32));
    float e = __expf(v - m), sum = e;
    for (int o = 16; o >= 1; o >>= 1) sum += __shfl_xor(sum, o, 32);
    gate[lane] = (lane < TT) ? (e / sum) : 0.f;
  }
  __syncthreads();

  // gated h (zero-padded to 32 rows)
  for (int i = tid; i < 32 * 136; i += 256) hg[i] = 0.f;
  __syncthreads();
  for (int i = tid; i < TT * 128; i += 256) {
    int r = i >> 7, c = i & 127;
    hg[r * 136 + c] = h_ln[((size_t)b * TT + r) * HH + c] * gate[r];
  }
  __syncthreads();

  // LayerNorm -> bf16 hn (pad rows zeroed)
  {
    const int sub = tid & 15;
    for (int pass = 0; pass < 2; ++pass) {
      int r = pass * 16 + (tid >> 4);
      float s = 0.f, ss = 0.f, vals[8];
#pragma unroll
      for (int j = 0; j < 8; ++j) {
        float v = hg[r * 136 + sub * 8 + j];
        vals[j] = v; s += v; ss += v * v;
      }
#pragma unroll
      for (int o = 8; o >= 1; o >>= 1) {
        s  += __shfl_xor(s, o, 16);
        ss += __shfl_xor(ss, o, 16);
      }
      float mean = s * (1.f / 128.f);
      float rstd = rsqrtf(ss * (1.f / 128.f) - mean * mean + LN_EPS);
#pragma unroll
      for (int j = 0; j < 8; ++j) {
        int c = sub * 8 + j;
        float v = (r < TT) ? ((vals[j] - mean) * rstd * n_g[c] + n_b[c]) : 0.f;
        hnb[r * 136 + c] = (__bf16)v;
      }
    }
  }
  __syncthreads();

  // q/k/v projections: 48 wave-tile tasks, bf16 WMMA
  for (int task = wave; task < 48; task += 8) {
    int mat = task >> 4, rem = task & 15;
    int nt = rem & 7, m0 = (rem >> 3) * 16;
    const __bf16* Wb  = (mat == 0) ? Wqb : (mat == 1) ? Wkb : Wvb;
    const float* bias = (mat == 0) ? bq  : (mat == 1) ? bk  : bv;
    float* dst        = (mat == 0) ? qs  : (mat == 1) ? ks  : vs;
    v8f acc = {};
    acc = wmma_bf16_tile(hnb + m0 * 136, 136, Wb + nt * 512, 8 * 512, 4, acc);
    float bia = bias[nt * 16 + lr];
#pragma unroll
    for (int i = 0; i < 8; ++i)
      dst[(m0 + i + 8 * hl) * 136 + nt * 16 + lr] = acc[i] + bia;
  }
  __syncthreads();

  // scores S = (q*scale) @ k^T  (f32 WMMA, transposed-B access)
  const float scale = 0.08838834764831845f;  // 1/sqrt(128)
  if (wave < 4) {
    int m0 = (wave >> 1) * 16, n0 = (wave & 1) * 16;
    v8f acc = {};
    acc = wmma_f32_tile(qs + m0 * 136, 136, ks + n0 * 136, 1, 136, 128, acc);
#pragma unroll
    for (int i = 0; i < 8; ++i) {
      int rr = m0 + i + 8 * hl, cc = n0 + lr;
      Ss[rr * 34 + cc] = (rr >= TT) ? 0.f
                        : ((cc >= TT) ? -3.0e38f : acc[i] * scale);
    }
  }
  __syncthreads();

  // row softmax over T (one wave per row)
  for (int r = wave; r < TT; r += 8) {
    float v = Ss[r * 34 + lane], m = v;
    for (int o = 16; o >= 1; o >>= 1) m = fmaxf(m, __shfl_xor(m, o, 32));
    float e = __expf(v - m), sum = e;
    for (int o = 16; o >= 1; o >>= 1) sum += __shfl_xor(sum, o, 32);
    Ss[r * 34 + lane] = e / sum;
  }
  __syncthreads();

  // attnv = P @ v (f32 WMMA, K=32) -> bf16
  for (int task = wave; task < 16; task += 8) {
    int m0 = (task >> 3) * 16, nt = task & 7;
    v8f acc = {};
    acc = wmma_f32_tile(Ss + m0 * 34, 34, vs + nt * 16, 136, 1, 32, acc);
#pragma unroll
    for (int i = 0; i < 8; ++i)
      avb[(m0 + i + 8 * hl) * 136 + nt * 16 + lr] = (__bf16)acc[i];
  }
  __syncthreads();

  // O = attnv @ Wo + bo + residual(hg)  -> bf16 hcur
  for (int task = wave; task < 16; task += 8) {
    int m0 = (task >> 3) * 16, nt = task & 7;
    v8f acc = {};
    acc = wmma_bf16_tile(avb + m0 * 136, 136, Wob + nt * 512, 8 * 512, 4, acc);
    float bia = bo[nt * 16 + lr];
#pragma unroll
    for (int i = 0; i < 8; ++i) {
      int rr = m0 + i + 8 * hl, cc = nt * 16 + lr;
      if (rr < TT)
        hcurb[((size_t)b * TT + rr) * HH + cc] =
            (__bf16)(acc[i] + bia + hg[rr * 136 + cc]);
    }
  }
}

// ---------------------------------------------------------------------------
// Kernel 3: out = hcur @ W2 + b2   [16,128]x[128,1024] per block.
// A-slab staged with gfx1250 async global->LDS copy (ASYNCcnt-tracked):
// 256 threads x 16B = entire 16x128 bf16 slab in one async instruction,
// per-lane LDS addresses implement the bank-padded stride-136 layout.
// ---------------------------------------------------------------------------
__global__ __launch_bounds__(256) void up_proj(
    const __bf16* __restrict__ hcurb, const __bf16* __restrict__ W2b,
    const float* __restrict__ b2, float* __restrict__ out) {
  __shared__ __align__(16) __bf16 hcb[16 * 136];
  const int tid = threadIdx.x;
  const int row0 = blockIdx.x * 16;
  {
    int r = tid >> 4, sub = tid & 15;   // 16 chunks of 16B per row
    const __bf16* g = hcurb + (size_t)(row0 + r) * HH + sub * 8;
    unsigned lds_off = (unsigned)(uintptr_t)&hcb[r * 136 + sub * 8];
    asm volatile("global_load_async_to_lds_b128 %0, %1, off"
                 :: "v"(lds_off), "v"(g)
                 : "memory");
    asm volatile("s_wait_asynccnt 0" ::: "memory");
  }
  __syncthreads();
  const int wave = tid >> 5, lane = tid & 31;
  const int hl = lane >> 4, lr = lane & 15;
  for (int j = 0; j < 8; ++j) {
    int nt = wave + 8 * j;  // 0..63 column tiles of D=1024
    v8f acc = {};
    acc = wmma_bf16_tile(hcb, 136, W2b + nt * 512, 64 * 512, 4, acc);
    int cc = nt * 16 + lr;
    float bia = b2[cc];
#pragma unroll
    for (int i = 0; i < 8; ++i)
      out[(size_t)(row0 + i + 8 * hl) * DD + cc] = acc[i] + bia;
  }
}

// ---------------------------------------------------------------------------
extern "C" void kernel_launch(void* const* d_in, const int* in_sizes, int n_in,
                              void* d_out, int out_size, void* d_ws, size_t ws_size,
                              hipStream_t stream) {
  const float* x    = (const float*)d_in[0];
  const float* W1   = (const float*)d_in[1];
  const float* b1   = (const float*)d_in[2];
  const float* ln_g = (const float*)d_in[3];
  const float* ln_b = (const float*)d_in[4];
  const float* T_W  = (const float*)d_in[5];
  const float* T_b  = (const float*)d_in[6];
  const float* Wq   = (const float*)d_in[7];
  const float* bq   = (const float*)d_in[8];
  const float* Wk   = (const float*)d_in[9];
  const float* bk   = (const float*)d_in[10];
  const float* Wv   = (const float*)d_in[11];
  const float* bvp  = (const float*)d_in[12];
  const float* Wo   = (const float*)d_in[13];
  const float* bo   = (const float*)d_in[14];
  const float* n_g  = (const float*)d_in[15];
  const float* n_b  = (const float*)d_in[16];
  const float* W2   = (const float*)d_in[17];
  const float* b2   = (const float*)d_in[18];
  float* out = (float*)d_out;

  char* ws = (char*)d_ws;
  size_t off = 0;
  auto alloc = [&](size_t bytes) -> void* {
    void* p = ws + off;
    off = (off + bytes + 255) & ~(size_t)255;
    return p;
  };
  __bf16* W1b = (__bf16*)alloc((size_t)DD * HH * 2);
  __bf16* W2b = (__bf16*)alloc((size_t)HH * DD * 2);
  __bf16* Wqb = (__bf16*)alloc((size_t)HH * HH * 2);
  __bf16* Wkb = (__bf16*)alloc((size_t)HH * HH * 2);
  __bf16* Wvb = (__bf16*)alloc((size_t)HH * HH * 2);
  __bf16* Wob = (__bf16*)alloc((size_t)HH * HH * 2);
  float*  h_ln  = (float*)alloc((size_t)BT * HH * 4);
  float*  gmean = (float*)alloc((size_t)BT * 4);
  __bf16* hcurb = (__bf16*)alloc((size_t)BT * HH * 2);

  swizzle_w<<<(DD * HH + 255) / 256, 256, 0, stream>>>(W1, W1b, DD, HH);
  swizzle_w<<<(HH * DD + 255) / 256, 256, 0, stream>>>(W2, W2b, HH, DD);
  swizzle_w<<<(HH * HH + 255) / 256, 256, 0, stream>>>(Wq, Wqb, HH, HH);
  swizzle_w<<<(HH * HH + 255) / 256, 256, 0, stream>>>(Wk, Wkb, HH, HH);
  swizzle_w<<<(HH * HH + 255) / 256, 256, 0, stream>>>(Wv, Wvb, HH, HH);
  swizzle_w<<<(HH * HH + 255) / 256, 256, 0, stream>>>(Wo, Wob, HH, HH);

  down_proj_ln<<<NBLK, 256, 0, stream>>>(x, W1b, b1, ln_g, ln_b, h_ln, gmean);
  batch_attn<<<BB, 256, 0, stream>>>(h_ln, gmean, T_W, T_b,
                                     Wqb, bq, Wkb, bk, Wvb, bvp, Wob, bo,
                                     n_g, n_b, hcurb);
  up_proj<<<NBLK, 256, 0, stream>>>(hcurb, W2b, b2, out);
}
